// PairActorNetwork_18141941858763
// MI455X (gfx1250) — matrix operand order repaired and checked by reading, not compiled, not run on hardware
//
#include <hip/hip_runtime.h>
#include <hip/hip_bf16.h>

#define Bb 8
#define Nn 1000
#define Aa 200
#define Pp 100000
#define Ee 64

typedef __attribute__((ext_vector_type(16))) _Float16 v16h;
typedef __attribute__((ext_vector_type(8)))  _Float16 v8h;
typedef __attribute__((ext_vector_type(8)))  float    v8f;

// Static device scratch (avoids depending on ws_size); fully rewritten each call.
// Embeddings stored directly in f16: that is the precision the WMMA consumes,
// so nothing is lost vs converting at A-build time, and it halves load bytes
// and removes all v_cvt_pk_f16_f32 from the hot loop.
__device__ _Float16 g_node[Bb * Nn * Ee];
__device__ _Float16 g_anch[Bb * Aa * Ee];
__device__ int      g_rowany[Bb];

// ---------------------------------------------------------------------------
// Kernel 1: tiny 2-layer MLP embeddings (fp32 math, f16 store — negligible
// cost). dst == 0 -> g_node, dst == 1 -> g_anch. blockDim = 64, 1 row/block.
// ---------------------------------------------------------------------------
__global__ void embed_kernel(const float* __restrict__ X,
                             const float* __restrict__ W1, const float* __restrict__ b1,
                             const float* __restrict__ W2, const float* __restrict__ b2,
                             int D, int dst) {
    __shared__ float xs[16];
    __shared__ float hs[Ee];
    const int r = blockIdx.x;
    const int j = threadIdx.x;
    if (j < D) xs[j] = X[(size_t)r * D + j];
    __syncthreads();
    float h = b1[j];
    for (int k = 0; k < D; ++k) h = fmaf(xs[k], W1[k * Ee + j], h);
    hs[j] = fmaxf(h, 0.0f);
    __syncthreads();
    float o = b2[j];
    for (int k = 0; k < Ee; ++k) o = fmaf(hs[k], W2[k * Ee + j], o);
    _Float16* out = dst ? g_anch : g_node;
    out[(size_t)r * Ee + j] = (_Float16)o;
}

// ---------------------------------------------------------------------------
// Kernel 2: row_any = any(mask[b, :]) per batch.  grid = B, block = 256.
// ---------------------------------------------------------------------------
__global__ void rowany_kernel(const int* __restrict__ mask) {
    __shared__ int s[256];
    const int b = blockIdx.x, t = threadIdx.x;
    int any = 0;
    for (int i = t; i < Pp; i += 256) any |= (mask[(size_t)b * Pp + i] != 0);
    s[t] = any;
    __syncthreads();
    for (int off = 128; off; off >>= 1) {
        if (t < off) s[t] |= s[t + off];
        __syncthreads();
    }
    if (t == 0) g_rowany[b] = s[0];
}

// ---------------------------------------------------------------------------
// Kernel 3: pair MLP on WMMA.  One wave32 per 16-pair tile.
// hidden = relu(pair_feat(192) @ Wp1 + bp1);  logit = hidden @ Wp2 + bp2
// ---------------------------------------------------------------------------
__global__ void __launch_bounds__(256)
pair_kernel(const int* __restrict__ pairs, const int* __restrict__ mask,
            const float* __restrict__ Wp1, const float* __restrict__ bp1,
            const float* __restrict__ Wp2, const float* __restrict__ bp2,
            float* __restrict__ out) {
    // Wp1 packed into the B-matrix (32x16, f16) wave layout: 24 fragments
    // (6 K-chunks x 4 N-tiles) x 32 lanes x 16 halves = 24 KB of LDS.
    // (The compiler hoists the 24 loop-invariant fragment loads into VGPRs,
    //  so LDS is only read once per wave — intended.)
    __shared__ __attribute__((aligned(32))) _Float16 Bl[24 * 32 * 16];

    const int tid = threadIdx.x;
    for (int e = tid; e < 24 * 32 * 16; e += 256) {
        const int j    = e & 15;          // half index within lane fragment
        const int lane = (e >> 4) & 31;
        const int frag = e >> 9;          // 0..23
        const int c  = frag >> 2;         // K-chunk
        const int nn = frag & 3;          // N-tile
        const int col = (nn << 4) + (lane & 15);
        const int k   = 32 * c + ((j & 8) << 1) + ((lane >> 4) << 3) + (j & 7);   // B layout
        Bl[e] = (_Float16)Wp1[k * Ee + col];
    }
    __syncthreads();

    const int lane = tid & 31;
    const int lmod = lane & 15;

    // Per-lane output-layer constants: column = 16*nn + lmod
    float bias1v[4], w2v[4];
#pragma unroll
    for (int nn = 0; nn < 4; ++nn) {
        bias1v[nn] = bp1[nn * 16 + lmod];
        w2v[nn]    = Wp2[nn * 16 + lmod];
    }
    const float b2s = bp2[0];

    const int NT      = (Bb * Pp) / 16;          // 50000 tiles
    const int wid     = blockIdx.x * 8 + (tid >> 5);
    const int wstride = gridDim.x * 8;

    for (int t = wid; t < NT; t += wstride) {
        const int row0 = t * 16;
        const int b    = row0 / Pp;              // tiles never straddle batches
        const int p0   = row0 - b * Pp;
        const int p    = p0 + lmod;              // this lane's pair (row M=lmod)
        const int ni   = pairs[2 * p];
        const int ai   = pairs[2 * p + 1];
        // Rows viewed as 8 aligned groups of 8 halves (16 B each).
        const v8h* nrow8 = (const v8h*)(g_node + (size_t)(b * Nn + ni) * Ee);
        const v8h* arow8 = (const v8h*)(g_anch + (size_t)(b * Aa + ai) * Ee);
        const int lo = (lane < 16) ? 0 : 1;      // A-layout K sub-offset /8

        v8f acc0 = {}, acc1 = {}, acc2 = {}, acc3 = {};

#pragma unroll
        for (int c = 0; c < 6; ++c) {
            // This lane's K runs (in 8-half groups): g1 = 4*(c&1)+lo, g2 = g1+2
            const int g1 = (c & 1) * 4 + lo;
            const int g2 = g1 + 2;
            v8h alo, ahi;
            if (c < 2) {                         // node section (K 0..63)
                alo = nrow8[g1];
                ahi = nrow8[g2];
            } else if (c < 4) {                  // anchor section (K 64..127)
                alo = arow8[g1];
                ahi = arow8[g2];
            } else {                             // node*anchor (K 128..191)
                alo = nrow8[g1] * arow8[g1];     // v_pk_mul_f16
                ahi = nrow8[g2] * arow8[g2];
            }
            const v16h a = __builtin_shufflevector(alo, ahi,
                0, 1, 2, 3, 4, 5, 6, 7, 8, 9, 10, 11, 12, 13, 14, 15);

            const v16h bf0 = *(const v16h*)&Bl[((c * 4 + 0) * 32 + lane) * 16];
            const v16h bf1 = *(const v16h*)&Bl[((c * 4 + 1) * 32 + lane) * 16];
            const v16h bf2 = *(const v16h*)&Bl[((c * 4 + 2) * 32 + lane) * 16];
            const v16h bf3 = *(const v16h*)&Bl[((c * 4 + 3) * 32 + lane) * 16];

            acc0 = __builtin_amdgcn_wmma_f32_16x16x32_f16(false, a, false, bf0,
                                                          (short)0, acc0, false, false);
            acc1 = __builtin_amdgcn_wmma_f32_16x16x32_f16(false, a, false, bf1,
                                                          (short)0, acc1, false, false);
            acc2 = __builtin_amdgcn_wmma_f32_16x16x32_f16(false, a, false, bf2,
                                                          (short)0, acc2, false, false);
            acc3 = __builtin_amdgcn_wmma_f32_16x16x32_f16(false, a, false, bf3,
                                                          (short)0, acc3, false, false);
        }

        // Fused bias + ReLU + output layer.  C layout: lane = column (per
        // half), VGPR i = row (M = i for lanes 0-15, M = i+8 for 16-31).
        float partial[8];
#pragma unroll
        for (int i = 0; i < 8; ++i) {
            float s = 0.0f;
            s = fmaf(fmaxf(acc0[i] + bias1v[0], 0.0f), w2v[0], s);
            s = fmaf(fmaxf(acc1[i] + bias1v[1], 0.0f), w2v[1], s);
            s = fmaf(fmaxf(acc2[i] + bias1v[2], 0.0f), w2v[2], s);
            s = fmaf(fmaxf(acc3[i] + bias1v[3], 0.0f), w2v[3], s);
            partial[i] = s;
        }
        // Sum over the 16 columns of each half-wave (xor masks < 16 keep the
        // two halves independent).
#pragma unroll
        for (int m = 1; m <= 8; m <<= 1) {
#pragma unroll
            for (int i = 0; i < 8; ++i)
                partial[i] += __shfl_xor(partial[i], m, 32);
        }

        if (lmod == 0) {                 // lanes 0 (rows 0-7) and 16 (rows 8-15)
            const int mbase = (lane >> 4) * 8;
#pragma unroll
            for (int i = 0; i < 8; ++i) {
                const int pr = p0 + mbase + i;
                const size_t oidx = (size_t)b * Pp + pr;
                const float v = partial[i] + b2s;
                const bool keep = (mask[oidx] != 0) || (g_rowany[b] == 0);
                out[oidx] = keep ? v : -1.0e9f;
            }
        }
    }
}

// ---------------------------------------------------------------------------
extern "C" void kernel_launch(void* const* d_in, const int* in_sizes, int n_in,
                              void* d_out, int out_size, void* d_ws, size_t ws_size,
                              hipStream_t stream) {
    (void)in_sizes; (void)n_in; (void)d_ws; (void)ws_size; (void)out_size;

    const float* node_features   = (const float*)d_in[0];
    const float* anchor_features = (const float*)d_in[1];
    const int*   action_mask     = (const int*)d_in[2];
    const int*   action_pairs    = (const int*)d_in[3];
    const float* Wn1 = (const float*)d_in[4];
    const float* bn1 = (const float*)d_in[5];
    const float* Wn2 = (const float*)d_in[6];
    const float* bn2 = (const float*)d_in[7];
    const float* Wa1 = (const float*)d_in[8];
    const float* ba1 = (const float*)d_in[9];
    const float* Wa2 = (const float*)d_in[10];
    const float* ba2 = (const float*)d_in[11];
    const float* Wp1 = (const float*)d_in[12];
    const float* bp1 = (const float*)d_in[13];
    const float* Wp2 = (const float*)d_in[14];
    const float* bp2 = (const float*)d_in[15];
    float* out = (float*)d_out;

    // 1) Embedding MLPs (fp32 math, f16 store; negligible next to pair GEMM).
    embed_kernel<<<Bb * Nn, 64, 0, stream>>>(node_features, Wn1, bn1, Wn2, bn2, 16, 0);
    embed_kernel<<<Bb * Aa, 64, 0, stream>>>(anchor_features, Wa1, ba1, Wa2, ba2, 8, 1);

    // 2) Per-batch any(mask).
    rowany_kernel<<<Bb, 256, 0, stream>>>(action_mask);

    // 3) WMMA pair MLP + fused output layer + masking.
    pair_kernel<<<1024, 256, 0, stream>>>(action_pairs, action_mask,
                                          Wp1, bp1, Wp2, bp2, out);
}
